// ConvLif_69776038690975
// MI455X (gfx1250) — matrix operand order repaired and checked by reading, not compiled, not run on hardware
//
#include <hip/hip_runtime.h>

typedef __attribute__((ext_vector_type(16))) __bf16 v16bf;
typedef __attribute__((ext_vector_type(8)))  __bf16 v8bf;
typedef __attribute__((ext_vector_type(8)))  float  v8f;

#define T_DIM 4
#define B_DIM 8
#define C_IN  64
#define C_OUT 128
#define HW    64
#define K_TOT 576      // C_IN * 9
#define KT_STEPS 18    // 576 / 32
#define BS_STRIDE 40   // elements per (kt,n) row: 32 data + 8 pad -> 80B, 16B aligned
#define XR_STRIDE 68   // Xraw row stride: data at [2..65], pads at [1] and [66]

__device__ __forceinline__ unsigned short f2bf(float f) {
    unsigned int u = __builtin_bit_cast(unsigned int, f);
    unsigned int r = (u + 0x7FFFu + ((u >> 16) & 1u)) >> 16;  // round-nearest-even
    return (unsigned short)r;
}

// ---------------------------------------------------------------------------
// Pack conv weights [128][64][3][3] f32 into per-lane bf16 WMMA A-fragments:
// packedA[((kt*8 + mt)*32 + lane)*16 + e]   (CDNA5 16-bit A 16x32 layout)
// ---------------------------------------------------------------------------
__global__ void pack_weights_kernel(const float* __restrict__ conv_w,
                                    unsigned short* __restrict__ packedA) {
    const int blk  = blockIdx.x;        // kt*8 + mt, 144 blocks
    const int lane = threadIdx.x;       // 32 threads
    const int kt = blk >> 3;
    const int mt = blk & 7;
    const int khalf = lane >> 4;
    const int row = mt * 16 + (lane & 15);   // Cout row

    unsigned short vals[16];
#pragma unroll
    for (int e = 0; e < 16; ++e) {
        const int j = e >> 1, p = e & 1;
        const int koff = (j < 4 ? 2 * j : 16 + 2 * (j - 4)) + 8 * khalf + p;
        const int kk = kt * 32 + koff;       // 0..575
        const int cin = kk / 9;
        const int r = kk - cin * 9;
        const int kh = r / 3;
        const int kw = r - kh * 3;
        vals[e] = f2bf(conv_w[((row * C_IN + cin) * 3 + kh) * 3 + kw]);
    }
    unsigned short* dst = packedA + (size_t)((blk * 32) + lane) * 16;
#pragma unroll
    for (int e = 0; e < 16; ++e) dst[e] = vals[e];
}

// ---------------------------------------------------------------------------
// Fused conv3x3 (implicit GEMM, WMMA bf16) + bias + LayerNorm(C) + LIF.
// One WG per (b, h). Two-phase im2col staging with a precomputed offset
// table (k -> Xraw/Bs byte offsets, affine in n) kills per-iteration div
// chains. Accumulators + LIF state stay in registers across timesteps.
// ---------------------------------------------------------------------------
__global__ __launch_bounds__(256) void conv_ln_lif_kernel(
    const float* __restrict__ x,                 // [T,B,Cin,H,W] f32
    const unsigned short* __restrict__ packedA,  // bf16 A fragments
    const float* __restrict__ conv_b,
    const float* __restrict__ ln_w,
    const float* __restrict__ ln_b,
    float* __restrict__ out)                     // [T,B,Cout,H,W] f32 spikes
{
    __shared__ unsigned short Bs[KT_STEPS * 64 * BS_STRIDE];  // 90 KB
    __shared__ unsigned short Xraw[3 * C_IN * XR_STRIDE];     // 25.5 KB
    __shared__ uint4 tbl[K_TOT / 2];                          // 4.5 KB
    __shared__ float biasS[C_OUT], lnwS[C_OUT], lnbS[C_OUT];
    __shared__ float redSum[8 * 32], redSq[8 * 32];
    __shared__ float muS[64], rvS[64];

    const int tid = threadIdx.x;
    const int b = blockIdx.x >> 6;
    const int h = blockIdx.x & 63;

    for (int i = tid; i < C_OUT; i += 256) {
        biasS[i] = conv_b[i]; lnwS[i] = ln_w[i]; lnbS[i] = ln_b[i];
    }

    // ---- one-time k -> offset table (byte offsets, add n-term at use) ----
    for (int k2 = tid; k2 < K_TOT / 2; k2 += 256) {
        const int k = k2 * 2;
        unsigned int off[2];
#pragma unroll
        for (int q = 0; q < 2; ++q) {
            const int kk = k + q;
            const int cin = kk / 9;
            const int r = kk - cin * 9;
            const int kh = r / 3;
            const int kw = r - kh * 3;
            // Xraw element index for column n is  (kh*64+cin)*68 + (n+kw+1)
            off[q] = (unsigned int)(((kh * C_IN + cin) * XR_STRIDE + kw + 1) * 2);
        }
        const unsigned int soff =
            (unsigned int)((((k >> 5) * 64) * BS_STRIDE + (k & 31)) * 2);  // + n*80
        tbl[k2] = make_uint4(off[0], off[1], soff, 0);
    }

    const int lane  = tid & 31;
    const int wv    = tid >> 5;          // 0..7
    const int nt    = wv & 3;            // N subtile (16 cols)
    const int mhalf = wv >> 2;           // Cout rows 0-63 / 64-127
    const int n0    = nt * 16 + (lane & 15);
    const int khalf = lane >> 4;

    const unsigned short* apL = packedA + (size_t)(mhalf * 4) * 512 + (size_t)lane * 16;
    const unsigned short* bpL = &Bs[n0 * BS_STRIDE + khalf * 8];

    float vm[4][8];                      // LIF membrane potential (registers)
#pragma unroll
    for (int i = 0; i < 4; ++i)
#pragma unroll
        for (int r = 0; r < 8; ++r) vm[i][r] = 0.0f;

    for (int t = 0; t < T_DIM; ++t) {
        __syncthreads();   // prev iter consumed Bs/red; also fences tbl at t=0

        // ---- phase 1: stage raw halo rows (bf16 pairs, aligned stores) ---
        for (int rr = tid; rr < 3 * C_IN; rr += 256) {
            Xraw[rr * XR_STRIDE + 1]  = 0;
            Xraw[rr * XR_STRIDE + 66] = 0;
        }
        for (int idx = tid; idx < 3 * C_IN * 32; idx += 256) {
            const int row = idx >> 5;            // dh*64 + cin
            const int wp  = idx & 31;            // pair of w
            const int dh  = row >> 6;
            const int cin = row & 63;
            const int hh  = h + dh - 1;
            float2 xv = make_float2(0.0f, 0.0f);
            if ((unsigned)hh < 64u)
                xv = *reinterpret_cast<const float2*>(
                    &x[((((size_t)(t * B_DIM + b) * C_IN + cin) * HW + hh) << 6) + wp * 2]);
            const unsigned int packed =
                (unsigned int)f2bf(xv.x) | ((unsigned int)f2bf(xv.y) << 16);
            *reinterpret_cast<unsigned int*>(&Xraw[row * XR_STRIDE + 2 + wp * 2]) = packed;
        }
        __syncthreads();

        // ---- phase 2: table-driven transpose into B-fragment layout ------
        {
            const char* xrawB = reinterpret_cast<const char*>(Xraw);
            char* bsB = reinterpret_cast<char*>(Bs);
            const int n = tid & 63;
            const int k2base = tid >> 6;
#pragma unroll 4
            for (int s = 0; s < (K_TOT / 2) / 4; ++s) {
                const int k2 = k2base + s * 4;
                const uint4 e = tbl[k2];
                const unsigned short lo =
                    *reinterpret_cast<const unsigned short*>(xrawB + e.x + 2 * n);
                const unsigned short hi =
                    *reinterpret_cast<const unsigned short*>(xrawB + e.y + 2 * n);
                *reinterpret_cast<unsigned int*>(bsB + e.z + 80 * n) =
                    (unsigned int)lo | ((unsigned int)hi << 16);
            }
        }
        __syncthreads();

        // ---- GEMM: acc = W(128x576) * patches(576x64) --------------------
        v8f acc[4];
#pragma unroll
        for (int i = 0; i < 4; ++i) acc[i] = (v8f){0,0,0,0,0,0,0,0};

#define LOAD_A(kt, A)                                                          \
        _Pragma("unroll")                                                      \
        for (int i = 0; i < 4; ++i)                                            \
            A[i] = *reinterpret_cast<const v16bf*>(apL + (size_t)((kt) * 8 + i) * 512);

#define LOAD_B(kt, Bv)                                                         \
        {                                                                      \
            union { v16bf v; v8bf hh[2]; } u_;                                 \
            const unsigned short* p_ = bpL + (kt) * (64 * BS_STRIDE);          \
            u_.hh[0] = *reinterpret_cast<const v8bf*>(p_);                     \
            u_.hh[1] = *reinterpret_cast<const v8bf*>(p_ + 16);                \
            Bv = u_.v;                                                         \
        }

#define WMMA4(A, Bv)                                                           \
        _Pragma("unroll")                                                      \
        for (int i = 0; i < 4; ++i)                                            \
            acc[i] = __builtin_amdgcn_wmma_f32_16x16x32_bf16(                  \
                false, A[i], false, Bv, (short)0, acc[i], false, false);

        v16bf A0[4], A1[4];
        v16bf B0, B1;
        LOAD_A(0, A0); LOAD_B(0, B0);
#pragma unroll 1
        for (int kt = 0; kt < KT_STEPS; kt += 2) {
            LOAD_A(kt + 1, A1); LOAD_B(kt + 1, B1);
            WMMA4(A0, B0);
            if (kt + 2 < KT_STEPS) { LOAD_A(kt + 2, A0); LOAD_B(kt + 2, B0); }
            WMMA4(A1, B1);
        }

        // ---- add bias in-place; per-lane LN partials for column n0 -------
        float s = 0.0f, sq = 0.0f;
#pragma unroll
        for (int i = 0; i < 4; ++i) {
            const int mbase = (mhalf * 4 + i) * 16 + khalf * 8;
#pragma unroll
            for (int r = 0; r < 8; ++r) {
                float y = acc[i][r] + biasS[mbase + r];
                acc[i][r] = y;
                s += y; sq += y * y;
            }
        }
        redSum[wv * 32 + lane] = s;
        redSq[wv * 32 + lane]  = sq;
        __syncthreads();

        if (tid < 64) {   // one thread per column: combine 4 partials
            const int col = tid;
            const int cnt = col >> 4;     // nt of that column
            const int li  = col & 15;
            float ts = 0.0f, tq = 0.0f;
#pragma unroll
            for (int mh = 0; mh < 2; ++mh)
#pragma unroll
                for (int kh2 = 0; kh2 < 2; ++kh2) {
                    const int idx = (mh * 4 + cnt) * 32 + kh2 * 16 + li;
                    ts += redSum[idx]; tq += redSq[idx];
                }
            const float mu = ts * (1.0f / 128.0f);
            const float var = tq * (1.0f / 128.0f) - mu * mu;
            muS[col] = mu;
            rvS[col] = rsqrtf(var + 1e-5f);
        }
        __syncthreads();

        // ---- normalize + LIF (state in registers) + store spikes ---------
        const float mu = muS[n0], rv = rvS[n0];
        const size_t obase = ((size_t)(t * B_DIM + b) * C_OUT) * (HW * HW)
                           + (size_t)h * HW + n0;
#pragma unroll
        for (int i = 0; i < 4; ++i) {
            const int mbase = (mhalf * 4 + i) * 16 + khalf * 8;
#pragma unroll
            for (int r = 0; r < 8; ++r) {
                const int m = mbase + r;
                const float y = (acc[i][r] - mu) * rv * lnwS[m] + lnbS[m];
                float v = 0.5f * (vm[i][r] + y);       // v += (y - v)/TAU, TAU=2
                const float spike = (v >= 1.0f) ? 1.0f : 0.0f;
                vm[i][r] = (v >= 1.0f) ? 0.0f : v;     // hard reset to V_RESET=0
                out[obase + (size_t)m * (HW * HW)] = spike;
            }
        }
    }
}

extern "C" void kernel_launch(void* const* d_in, const int* in_sizes, int n_in,
                              void* d_out, int out_size, void* d_ws, size_t ws_size,
                              hipStream_t stream) {
    (void)in_sizes; (void)n_in; (void)out_size; (void)ws_size;
    const float* x      = (const float*)d_in[0];
    const float* conv_w = (const float*)d_in[1];
    const float* conv_b = (const float*)d_in[2];
    const float* ln_w   = (const float*)d_in[3];
    const float* ln_b   = (const float*)d_in[4];
    float* out = (float*)d_out;
    unsigned short* packedA = (unsigned short*)d_ws;   // 147456 B

    pack_weights_kernel<<<KT_STEPS * 8, 32, 0, stream>>>(conv_w, packedA);
    conv_ln_lif_kernel<<<B_DIM * HW, 256, 0, stream>>>(x, packedA, conv_b, ln_w, ln_b, out);
}
